// BertGNNClassifier_87557203296773
// MI455X (gfx1250) — matrix operand order repaired and checked by reading, not compiled, not run on hardware
//
#include <hip/hip_runtime.h>
#include <stdint.h>

#define IN_DIM 768
#define HID 128
#define NUM_CLASSES 32

typedef __attribute__((ext_vector_type(16))) __bf16 v16bf;
typedef __attribute__((ext_vector_type(8)))  float  v8f;

union BFrag {
    v16bf v;
    uint4 q[2];
    unsigned short u[16];
};

__device__ __forceinline__ unsigned short f2bf(float f) {
    // round-to-nearest-even f32 -> bf16
    unsigned u = __float_as_uint(f);
    unsigned r = u + 0x7FFFu + ((u >> 16) & 1u);
    return (unsigned short)(r >> 16);
}

// ---------------------------------------------------------------------------
// GEMM: C[M][Ntot] = A[M][K] (f32, converted to bf16 on the fly) x Wt (bf16 [Ntot][K])
// One wave computes a 16-row x (NT*16)-col strip via v_wmma_f32_16x16x32_bf16.
// blockIdx.y selects the column group (colBase = blockIdx.y * NT * 16).
// NT=4 keeps accumulators (32 VGPRs) + pipelined fragments spill-free.
// K must be a multiple of 32.
// ---------------------------------------------------------------------------
template <int NT>
__global__ void gemm_wmma_bf16(const float* __restrict__ A,
                               const unsigned short* __restrict__ Wt,
                               const float* __restrict__ bias,  // may be null
                               float* __restrict__ C,
                               int M, int K, int Ntot) {
    const int lane = threadIdx.x & 31;
    const int wave = threadIdx.x >> 5;
    const int mtile = blockIdx.x * (blockDim.x >> 5) + wave;
    if (mtile * 16 >= M) return;  // wave-uniform guard (EXEC stays all-ones)

    const int colBase = blockIdx.y * (NT * 16);
    const int half = lane >> 4;   // which 16-lane half of the wave
    const int m    = lane & 15;   // row (A) / column (B) index within tile
    const size_t arow = (size_t)(mtile * 16 + m) * (size_t)K;

    v8f acc[NT] = {};

    for (int k0 = 0; k0 < K; k0 += 32) {
        // ---- A fragment: element e -> K = (e/8)*16 + half*8 + e%8, row = m
        BFrag af;
        const float* ap = A + arow + k0 + half * 8;
        float4 a0 = *(const float4*)(ap);
        float4 a1 = *(const float4*)(ap + 4);
        float4 a2 = *(const float4*)(ap + 16);
        float4 a3 = *(const float4*)(ap + 20);
        af.u[0]  = f2bf(a0.x); af.u[1]  = f2bf(a0.y);
        af.u[2]  = f2bf(a0.z); af.u[3]  = f2bf(a0.w);
        af.u[4]  = f2bf(a1.x); af.u[5]  = f2bf(a1.y);
        af.u[6]  = f2bf(a1.z); af.u[7]  = f2bf(a1.w);
        af.u[8]  = f2bf(a2.x); af.u[9]  = f2bf(a2.y);
        af.u[10] = f2bf(a2.z); af.u[11] = f2bf(a2.w);
        af.u[12] = f2bf(a3.x); af.u[13] = f2bf(a3.y);
        af.u[14] = f2bf(a3.z); af.u[15] = f2bf(a3.w);

#pragma unroll
        for (int t = 0; t < NT; ++t) {
            // ---- B fragment: element e -> K = (e/8)*16 + half*8 + e%8, col = colBase + t*16 + m
            BFrag bfr;
            const uint4* bp = (const uint4*)(Wt +
                (size_t)(colBase + t * 16 + m) * (size_t)K + (size_t)(k0 + half * 8));
            bfr.q[0] = bp[0];
            bfr.q[1] = bp[2];
            acc[t] = __builtin_amdgcn_wmma_f32_16x16x32_bf16(
                false, af.v, false, bfr.v, (short)0, acc[t], false, false);
        }
    }

    // ---- epilogue: acc[t][r] -> C[mtile*16 + half*8 + r][colBase + t*16 + m]
    const int orow = mtile * 16 + half * 8;
#pragma unroll
    for (int t = 0; t < NT; ++t) {
        const int col = colBase + t * 16 + m;
        const float b = bias ? bias[col] : 0.0f;
#pragma unroll
        for (int r = 0; r < 8; ++r) {
            C[(size_t)(orow + r) * (size_t)Ntot + col] = acc[t][r] + b;
        }
    }
}

// ---------------------------------------------------------------------------
// Weight transpose + bf16 convert: W[K][N] (f32) -> Wt[N][K] (bf16)
// ---------------------------------------------------------------------------
__global__ void wtrans_kernel(const float* __restrict__ W,
                              unsigned short* __restrict__ Wt,
                              int K, int N) {
    int g = blockIdx.x * blockDim.x + threadIdx.x;
    if (g >= K * N) return;
    int k = g / N, n = g % N;
    Wt[(size_t)n * K + k] = f2bf(W[g]);
}

// ---------------------------------------------------------------------------
// Degree / norm kernels
// ---------------------------------------------------------------------------
__global__ void deg_init_kernel(float* __restrict__ deg, int n) {
    int i = blockIdx.x * blockDim.x + threadIdx.x;
    if (i < n) deg[i] = 1.0f;  // self-loop contributes 1 to every node
}

__global__ void deg_edges_kernel(const int* __restrict__ dst,
                                 float* __restrict__ deg, int nE) {
    int e = blockIdx.x * blockDim.x + threadIdx.x;
    if (e < nE) atomicAdd(&deg[dst[e]], 1.0f);
}

__global__ void deg_to_dinv_kernel(float* __restrict__ deg, int n) {
    int i = blockIdx.x * blockDim.x + threadIdx.x;
    if (i < n) deg[i] = rsqrtf(deg[i]);  // deg >= 1 always (self-loops)
}

// ---------------------------------------------------------------------------
// Aggregation: out[i] = dinv[i]^2 * g[i]  (self-loop term, initializes out)
// then edge scatter: out[dst] += dinv[src]*dinv[dst] * g[src]
// 32 feature-quads (float4) per node/edge; HID = 128.
// ---------------------------------------------------------------------------
__global__ void agg_self_kernel(const float* __restrict__ g,
                                const float* __restrict__ dinv,
                                float* __restrict__ out, int n) {
    long long gid = (long long)blockIdx.x * blockDim.x + threadIdx.x;
    int i  = (int)(gid >> 5);
    int fq = (int)(gid & 31);
    if (i >= n) return;
    float w = dinv[i];
    w *= w;
    float4 v = ((const float4*)(g + (size_t)i * HID))[fq];
    ((float4*)(out + (size_t)i * HID))[fq] =
        make_float4(w * v.x, w * v.y, w * v.z, w * v.w);
}

__global__ void agg_edges_kernel(const int* __restrict__ src,
                                 const int* __restrict__ dst,
                                 const float* __restrict__ dinv,
                                 const float* __restrict__ g,
                                 float* __restrict__ out, int nE) {
    long long gid = (long long)blockIdx.x * blockDim.x + threadIdx.x;
    int e  = (int)(gid >> 5);
    int fq = (int)(gid & 31);
    if (e >= nE) return;
    int s = src[e], d = dst[e];
    float nm = dinv[s] * dinv[d];
    float4 v = ((const float4*)(g + (size_t)s * HID))[fq];
    float* o = out + (size_t)d * HID + (size_t)fq * 4;
    atomicAdd(o + 0, nm * v.x);
    atomicAdd(o + 1, nm * v.y);
    atomicAdd(o + 2, nm * v.z);
    atomicAdd(o + 3, nm * v.w);
}

__global__ void bias_relu_kernel(float* __restrict__ h,
                                 const float* __restrict__ b, int n) {
    long long gid = (long long)blockIdx.x * blockDim.x + threadIdx.x;
    int i  = (int)(gid >> 5);
    int fq = (int)(gid & 31);
    if (i >= n) return;
    int f = fq * 4;
    float4 v = ((float4*)(h + (size_t)i * HID))[fq];
    v.x = fmaxf(v.x + b[f + 0], 0.0f);
    v.y = fmaxf(v.y + b[f + 1], 0.0f);
    v.z = fmaxf(v.z + b[f + 2], 0.0f);
    v.w = fmaxf(v.w + b[f + 3], 0.0f);
    ((float4*)(h + (size_t)i * HID))[fq] = v;
}

// ---------------------------------------------------------------------------
static inline size_t align256(size_t x) { return (x + 255) & ~(size_t)255; }
static inline int cdiv(long long a, long long b) { return (int)((a + b - 1) / b); }

extern "C" void kernel_launch(void* const* d_in, const int* in_sizes, int n_in,
                              void* d_out, int out_size, void* d_ws, size_t ws_size,
                              hipStream_t stream) {
    const float* x   = (const float*)d_in[0];
    const int*   ei  = (const int*)d_in[1];
    const float* W1  = (const float*)d_in[2];
    const float* b1  = (const float*)d_in[3];
    const float* W2  = (const float*)d_in[4];
    const float* b2  = (const float*)d_in[5];
    const float* Wfc = (const float*)d_in[6];
    const float* bfc = (const float*)d_in[7];
    float* out = (float*)d_out;

    const int nNodes = in_sizes[0] / IN_DIM;   // 50000
    const int nE     = in_sizes[1] / 2;        // 1600000
    const int* src = ei;
    const int* dst = ei + nE;

    // workspace carve-up
    char* ws = (char*)d_ws;
    float* dinv = (float*)ws;                  ws += align256((size_t)nNodes * 4);
    float* bufA = (float*)ws;                  ws += align256((size_t)nNodes * HID * 4);
    float* bufB = (float*)ws;                  ws += align256((size_t)nNodes * HID * 4);
    unsigned short* Wt1  = (unsigned short*)ws; ws += align256((size_t)IN_DIM * HID * 2);
    unsigned short* Wt2  = (unsigned short*)ws; ws += align256((size_t)HID * HID * 2);
    unsigned short* Wtfc = (unsigned short*)ws; ws += align256((size_t)HID * NUM_CLASSES * 2);
    (void)ws_size; (void)n_in; (void)out_size;

    const int B = 256;

    // weight transpose + convert (tiny)
    wtrans_kernel<<<cdiv((long long)IN_DIM * HID, B), B, 0, stream>>>(W1, Wt1, IN_DIM, HID);
    wtrans_kernel<<<cdiv((long long)HID * HID, B), B, 0, stream>>>(W2, Wt2, HID, HID);
    wtrans_kernel<<<cdiv((long long)HID * NUM_CLASSES, B), B, 0, stream>>>(Wfc, Wtfc, HID, NUM_CLASSES);

    // degrees -> dinv (stored in-place)
    deg_init_kernel<<<cdiv(nNodes, B), B, 0, stream>>>(dinv, nNodes);
    deg_edges_kernel<<<cdiv(nE, B), B, 0, stream>>>(dst, dinv, nE);
    deg_to_dinv_kernel<<<cdiv(nNodes, B), B, 0, stream>>>(dinv, nNodes);

    const int mtiles   = cdiv(nNodes, 16);
    const int gemmGrid = cdiv(mtiles, 8);                // 8 waves / block
    const long long nodeFeat = (long long)nNodes * 32;   // node x float4-quad
    const long long edgeFeat = (long long)nE * 32;

    dim3 gHid(gemmGrid, HID / 64);          // NT=4 -> 64 cols per group
    dim3 gCls(gemmGrid, NUM_CLASSES / 32);  // NT=2 -> 32 cols per group

    // ---- layer 1: g = x @ W1 ; agg ; +b1 relu
    gemm_wmma_bf16<4><<<gHid, B, 0, stream>>>(x, Wt1, nullptr, bufA, nNodes, IN_DIM, HID);
    agg_self_kernel<<<cdiv(nodeFeat, B), B, 0, stream>>>(bufA, dinv, bufB, nNodes);
    agg_edges_kernel<<<cdiv(edgeFeat, B), B, 0, stream>>>(src, dst, dinv, bufA, bufB, nE);
    bias_relu_kernel<<<cdiv(nodeFeat, B), B, 0, stream>>>(bufB, b1, nNodes);

    // ---- layer 2: g = h1 @ W2 ; agg ; +b2 relu
    gemm_wmma_bf16<4><<<gHid, B, 0, stream>>>(bufB, Wt2, nullptr, bufA, nNodes, HID, HID);
    agg_self_kernel<<<cdiv(nodeFeat, B), B, 0, stream>>>(bufA, dinv, bufB, nNodes);
    agg_edges_kernel<<<cdiv(edgeFeat, B), B, 0, stream>>>(src, dst, dinv, bufA, bufB, nE);
    bias_relu_kernel<<<cdiv(nodeFeat, B), B, 0, stream>>>(bufB, b2, nNodes);

    // ---- classifier head: out = h2 @ Wfc + bfc
    gemm_wmma_bf16<2><<<gCls, B, 0, stream>>>(bufB, Wtfc, bfc, out, nNodes, HID, NUM_CLASSES);
}
